// Deprecated_Cartesian_similarity_43052752175450
// MI455X (gfx1250) — matrix operand order repaired and checked by reading, not compiled, not run on hardware
//
#include <hip/hip_runtime.h>
#include <hip/hip_bf16.h>
#include <math.h>
#include <stdint.h>

#define NROW 4096
#define DDIM 1024
#define TEMP_INV 20.0f
#define EPSN 1e-8f
#define BK 32
#define LDSROW 40   // padded LDS row stride in bf16 (80 B): bank-conflict-free ds_load_b128

typedef __bf16 bf16;
typedef __attribute__((ext_vector_type(16))) __bf16 v16bf;
typedef __attribute__((ext_vector_type(8)))  __bf16 v8bf;
typedef __attribute__((ext_vector_type(8)))  float  v8f;

// ---------------------------------------------------------------------------
// CDNA5 async global->LDS copy (ASYNCcnt-tracked, bypasses VGPRs).
// GV mode: per-lane LDS byte address in vdst, 64-bit global address in vaddr.
// ---------------------------------------------------------------------------
__device__ __forceinline__ void async_b128(uint32_t lds_addr, const void* gaddr) {
    asm volatile("global_load_async_to_lds_b128 %0, %1, off"
                 :: "v"(lds_addr), "v"(gaddr)
                 : "memory");
}
__device__ __forceinline__ void wait_async0() {
    asm volatile("s_wait_asynccnt 0x0" ::: "memory");
}

// ---------------------------------------------------------------------------
// Kernel 1: L2-normalize rows of a [4096,1024] f32 matrix -> bf16
// ---------------------------------------------------------------------------
__global__ __launch_bounds__(256)
void normalize_rows(const float* __restrict__ src, bf16* __restrict__ dst) {
    const int row  = blockIdx.x;
    const int tid  = threadIdx.x;
    const int lane = tid & 31;
    const int wave = tid >> 5;

    const float4 x = *(const float4*)(src + (size_t)row * DDIM + tid * 4);
    float s = x.x * x.x + x.y * x.y + x.z * x.z + x.w * x.w;
    #pragma unroll
    for (int off = 16; off; off >>= 1) s += __shfl_xor(s, off, 32);

    __shared__ float red[8];
    if (lane == 0) red[wave] = s;
    __syncthreads();
    float tot = 0.f;
    #pragma unroll
    for (int w = 0; w < 8; ++w) tot += red[w];

    const float inv = 1.0f / fmaxf(sqrtf(tot), EPSN);
    bf16* o = dst + (size_t)row * DDIM + tid * 4;
    o[0] = (bf16)(x.x * inv);
    o[1] = (bf16)(x.y * inv);
    o[2] = (bf16)(x.z * inv);
    o[3] = (bf16)(x.w * inv);
}

// ---------------------------------------------------------------------------
// Fragment assembly from LDS, per CDNA5 ISA wave32 VGPR layouts.
// A 16x32 bf16: lanes 0-15: K 0-7 | 16-23 ; lanes 16-31: K 8-15 | 24-31
// B 32x16 bf16: lanes 0-15: K 0-7 | 16-23 ; lanes 16-31: K 8-15 | 24-31
// ---------------------------------------------------------------------------
__device__ __forceinline__ v16bf frag_pair(const bf16* p, int second_off) {
    v8bf lo = *(const v8bf*)(p);
    v8bf hi = *(const v8bf*)(p + second_off);
    v16bf r;
    #pragma unroll
    for (int e = 0; e < 8; ++e) { r[e] = lo[e]; r[e + 8] = hi[e]; }
    return r;
}

// ---------------------------------------------------------------------------
// Kernel 2: sim = An x Bn^T, bf16 WMMA with async-LDS double buffering.
// Block 256 thr = 8 waves, block tile 128x128, wave tile 64x32 (4x2 WMMA).
// ---------------------------------------------------------------------------
__global__ __launch_bounds__(256)
void gemm_bf16_wmma(const bf16* __restrict__ A, const bf16* __restrict__ B,
                    float* __restrict__ C) {
    __shared__ alignas(16) bf16 Ash[2][128 * LDSROW];
    __shared__ alignas(16) bf16 Bsh[2][128 * LDSROW];

    const int tid  = threadIdx.x;
    const int lane = tid & 31;
    const int wave = tid >> 5;
    const int mw   = wave & 1;       // 2 waves along M (64 rows each)
    const int nw   = wave >> 1;      // 4 waves along N (32 cols each)
    const int bM   = blockIdx.x * 128;
    const int bN   = blockIdx.y * 128;
    const int r    = lane & 15;
    const int hi   = lane >> 4;

    // Staging: 128x32 bf16 tile = 512 x 16B chunks; 2 chunks/thread/tile.
    const int c0 = tid,        r0 = c0 >> 2, s0 = c0 & 3;
    const int c1 = tid + 256,  r1 = c1 >> 2, s1 = c1 & 3;

    auto stage = [&](int buf, int k) {
        async_b128((uint32_t)(uintptr_t)&Ash[buf][r0 * LDSROW + s0 * 8],
                   A + (size_t)(bM + r0) * DDIM + k + s0 * 8);
        async_b128((uint32_t)(uintptr_t)&Ash[buf][r1 * LDSROW + s1 * 8],
                   A + (size_t)(bM + r1) * DDIM + k + s1 * 8);
        async_b128((uint32_t)(uintptr_t)&Bsh[buf][r0 * LDSROW + s0 * 8],
                   B + (size_t)(bN + r0) * DDIM + k + s0 * 8);
        async_b128((uint32_t)(uintptr_t)&Bsh[buf][r1 * LDSROW + s1 * 8],
                   B + (size_t)(bN + r1) * DDIM + k + s1 * 8);
    };

    v8f acc[4][2] = {};

    stage(0, 0);
    #pragma unroll 2
    for (int kt = 0; kt < DDIM / BK; ++kt) {
        const int buf = kt & 1;
        wait_async0();
        __syncthreads();                      // staged tiles visible block-wide
        if (kt + 1 < DDIM / BK) stage(buf ^ 1, (kt + 1) * BK);

        v16bf af[4], bfr[2];
        #pragma unroll
        for (int mt = 0; mt < 4; ++mt)
            af[mt] = frag_pair(&Ash[buf][(mw * 64 + mt * 16 + r) * LDSROW + hi * 8], 16);
        #pragma unroll
        for (int nt = 0; nt < 2; ++nt)
            bfr[nt] = frag_pair(&Bsh[buf][(nw * 32 + nt * 16 + r) * LDSROW + hi * 8], 16);

        #pragma unroll
        for (int mt = 0; mt < 4; ++mt)
            #pragma unroll
            for (int nt = 0; nt < 2; ++nt)
                acc[mt][nt] = __builtin_amdgcn_wmma_f32_16x16x32_bf16(
                    false, af[mt], false, bfr[nt], (short)0, acc[mt][nt], false, false);

        __syncthreads();                      // all reads of buf done before rewrite
    }

    // C/D layout: lanes 0-15 -> N=lane, M=vgpr; lanes 16-31 -> M=8+vgpr.
    #pragma unroll
    for (int mt = 0; mt < 4; ++mt) {
        #pragma unroll
        for (int nt = 0; nt < 2; ++nt) {
            const int col     = bN + nw * 32 + nt * 16 + r;
            const int rowBase = bM + mw * 64 + mt * 16 + hi * 8;
            #pragma unroll
            for (int v = 0; v < 8; ++v)
                C[(size_t)(rowBase + v) * NROW + col] = acc[mt][nt][v];
        }
    }
}

// ---------------------------------------------------------------------------
// Kernel 3: y_true[i,j] = (label[i] == label[j])
// ---------------------------------------------------------------------------
__global__ __launch_bounds__(256)
void ytrue_kernel(const int* __restrict__ lab, int* __restrict__ yt) {
    const size_t idx = (size_t)blockIdx.x * 256 + threadIdx.x;
    const int i = (int)(idx >> 12);
    const int j = (int)(idx & (NROW - 1));
    yt[idx] = (lab[i] == lab[j]) ? 1 : 0;
}

// ---------------------------------------------------------------------------
// Kernel 4: per-row masked softmax statistics.
// rowval[i] = sum_j mask*logit / cnt - log(sum_j exp(logit))
// ---------------------------------------------------------------------------
__global__ __launch_bounds__(256)
void row_stats(const float* __restrict__ sim, const int* __restrict__ lab,
               float* __restrict__ rowval) {
    const int i    = blockIdx.x;
    const int tid  = threadIdx.x;
    const int lane = tid & 31;
    const int wave = tid >> 5;
    const int li   = lab[i];
    const float* rowp = sim + (size_t)i * NROW;

    float m = -3.402823466e38f;
    for (int j = tid; j < NROW; j += 256) m = fmaxf(m, rowp[j] * TEMP_INV);
    #pragma unroll
    for (int off = 16; off; off >>= 1) m = fmaxf(m, __shfl_xor(m, off, 32));

    __shared__ float redm[8];
    if (lane == 0) redm[wave] = m;
    __syncthreads();
    float rm = redm[0];
    #pragma unroll
    for (int w = 1; w < 8; ++w) rm = fmaxf(rm, redm[w]);

    float se = 0.f, ms = 0.f, cnt = 0.f;
    for (int j = tid; j < NROW; j += 256) {
        const float l = rowp[j] * TEMP_INV - rm;
        se += __expf(l);
        if (lab[j] == li) { ms += l; cnt += 1.f; }
    }
    #pragma unroll
    for (int off = 16; off; off >>= 1) {
        se  += __shfl_xor(se,  off, 32);
        ms  += __shfl_xor(ms,  off, 32);
        cnt += __shfl_xor(cnt, off, 32);
    }
    __shared__ float r1[8], r2[8], r3[8];
    if (lane == 0) { r1[wave] = se; r2[wave] = ms; r3[wave] = cnt; }
    __syncthreads();
    if (tid == 0) {
        float S = 0.f, M = 0.f, Cn = 0.f;
        #pragma unroll
        for (int w = 0; w < 8; ++w) { S += r1[w]; M += r2[w]; Cn += r3[w]; }
        rowval[i] = M / Cn - logf(S);
    }
}

// ---------------------------------------------------------------------------
// Kernel 5: loss = -mean(rowval)
// ---------------------------------------------------------------------------
__global__ __launch_bounds__(256)
void loss_reduce(const float* __restrict__ rowval, float* __restrict__ lossp) {
    const int tid = threadIdx.x;
    float s = 0.f;
    for (int j = tid; j < NROW; j += 256) s += rowval[j];
    #pragma unroll
    for (int off = 16; off; off >>= 1) s += __shfl_xor(s, off, 32);
    __shared__ float red[8];
    if ((tid & 31) == 0) red[tid >> 5] = s;
    __syncthreads();
    if (tid == 0) {
        float t = 0.f;
        #pragma unroll
        for (int w = 0; w < 8; ++w) t += red[w];
        *lossp = -t / (float)NROW;
    }
}

// ---------------------------------------------------------------------------
extern "C" void kernel_launch(void* const* d_in, const int* in_sizes, int n_in,
                              void* d_out, int out_size, void* d_ws, size_t ws_size,
                              hipStream_t stream) {
    const float* c   = (const float*)d_in[0];
    const float* ch  = (const float*)d_in[1];
    const int*   lab = (const int*)d_in[2];

    float* sim   = (float*)d_out;                       // [N,N] f32
    float* lossp = sim + (size_t)NROW * NROW;           // scalar
    int*   yt    = (int*)(lossp + 1);                   // [N,N] i32

    // bf16 scratch (16 MB) lives inside the 64 MB y_true region, 16-B aligned.
    // gemm finishes before ytrue_kernel overwrites it (stream-ordered).
    char* scratch = (char*)d_out + (size_t)NROW * NROW * sizeof(float) + 16;
    bf16* Abf = (bf16*)scratch;
    bf16* Bbf = Abf + (size_t)NROW * DDIM;

    float* rowval = (float*)d_ws;                       // 16 KB

    normalize_rows<<<NROW, 256, 0, stream>>>(c,  Abf);
    normalize_rows<<<NROW, 256, 0, stream>>>(ch, Bbf);

    dim3 g(NROW / 128, NROW / 128);
    gemm_bf16_wmma<<<g, 256, 0, stream>>>(Abf, Bbf, sim);

    ytrue_kernel<<<(NROW / 256) * NROW, 256, 0, stream>>>(lab, yt);
    row_stats<<<NROW, 256, 0, stream>>>(sim, lab, rowval);
    loss_reduce<<<1, 256, 0, stream>>>(rowval, lossp);
}
